// SAGE_73126113182159
// MI455X (gfx1250) — matrix operand order repaired and checked by reading, not compiled, not run on hardware
//
#include <hip/hip_runtime.h>
#include <math.h>

typedef __attribute__((ext_vector_type(16))) _Float16 v16h;
typedef __attribute__((ext_vector_type(8)))  float    v8f;
typedef __attribute__((ext_vector_type(2)))  _Float16 h2;
typedef __attribute__((ext_vector_type(4)))  _Float16 h4;
typedef int vi4 __attribute__((vector_size(16)));   // matches async builtin param

#define NNODES 50000
#define NEDGES 500000
#define DFEAT  512          // feature width of x / h / agg
#define KCAT   1024         // concatenated K = [agg | root]
#define BM     128
#define BN     128
#define BK     32

#if defined(__gfx1250__) && __has_builtin(__builtin_amdgcn_global_load_async_to_lds_b128)
#define HAVE_ASYNC_LDS 1
#else
#define HAVE_ASYNC_LDS 0
#endif

__device__ __forceinline__ void wait_async_zero() {
#if defined(__gfx1250__)
#if __has_builtin(__builtin_amdgcn_s_wait_asynccnt)
    __builtin_amdgcn_s_wait_asynccnt(0);
#else
    asm volatile("s_wait_asynccnt 0" ::: "memory");
#endif
#endif
}

// ---------------------------------------------------------------------------
// Pack [Wl | Wr] (each rows x 512, f32 row-major) into f16 [rows x 1024]
// ---------------------------------------------------------------------------
__global__ void sage_pack_w(const float* __restrict__ Wl,
                            const float* __restrict__ Wr,
                            _Float16* __restrict__ out, int rows) {
    int i = blockIdx.x * blockDim.x + threadIdx.x;
    int total = rows * KCAT;
    if (i >= total) return;
    int row = i >> 10;          // /1024
    int col = i & 1023;
    float v = (col < DFEAT) ? Wl[(size_t)row * DFEAT + col]
                            : Wr[(size_t)row * DFEAT + (col - DFEAT)];
    out[i] = (_Float16)v;
}

// ---------------------------------------------------------------------------
// Edge scatter: one wave32 per edge.  agg[dst] += feat[src]; deg[dst] += 1.
// agg (102 MB) is L2-resident on MI455X (192 MB L2) -> cheap f32 atomics.
// ---------------------------------------------------------------------------
__global__ void sage_scatter(const float* __restrict__ feat,
                             const long long* __restrict__ eidx,
                             float* __restrict__ agg,
                             float* __restrict__ deg, int E) {
    int wave = (blockIdx.x * blockDim.x + threadIdx.x) >> 5;
    int lane = threadIdx.x & 31;
    if (wave >= E) return;
    long long s = eidx[wave];
    long long d = eidx[(size_t)E + wave];
    const float4* srow = (const float4*)(feat + (size_t)s * DFEAT);
    float* drow = agg + (size_t)d * DFEAT;
#pragma unroll
    for (int q = lane; q < DFEAT / 4; q += 32) {
        float4 v = srow[q];
        atomicAdd(&drow[q * 4 + 0], v.x);
        atomicAdd(&drow[q * 4 + 1], v.y);
        atomicAdd(&drow[q * 4 + 2], v.z);
        atomicAdd(&drow[q * 4 + 3], v.w);
    }
    if (lane == 0) atomicAdd(&deg[d], 1.0f);
}

__global__ void sage_rdeg(const float* __restrict__ deg,
                          float* __restrict__ rdeg, int n) {
    int i = blockIdx.x * blockDim.x + threadIdx.x;
    if (i < n) rdeg[i] = 1.0f / fmaxf(deg[i], 1.0f);
}

// ---------------------------------------------------------------------------
// Fused GEMM:  out[M x Nout] = [agg*rdeg | root] (M x 1024) @ Wcat^T + bias
// Wcat is f16 [Nout x 1024] row-major.  WMMA f32_16x16x32_f16, f32 accum.
// Block: 256 threads = 8 waves (4M x 2N), tile 128x128, BK=32.
// Wave tile 32x64 = 2x4 WMMA accumulators -> 8 v_wmma per k-step.
// B-tile staged via ASYNCcnt-tracked global_load_async_to_lds_b128 (two
// unconditional DMA chunks per thread); A-tile staged through VGPRs
// (needs f32->f16 convert + rdeg mean scale, hoisted per row).
// ---------------------------------------------------------------------------
__global__ __launch_bounds__(256)
void sage_gemm(const float* __restrict__ agg,
               const float* __restrict__ rdeg,
               const float* __restrict__ root,
               const _Float16* __restrict__ Wcat,
               const float* __restrict__ bias,
               float* __restrict__ out,
               int M, int Nout, int do_relu) {
    __shared__ __align__(16) _Float16 As[BM][BK + 8];   // row stride 80 B
    __shared__ __align__(16) _Float16 Bs[BN][BK + 8];   // (16 B aligned rows)

    const int tid  = threadIdx.x;
    const int lane = tid & 31;
    const int wid  = tid >> 5;           // 0..7
    const int wm   = (wid & 3) * 32;     // wave M origin in block tile
    const int wn   = (wid >> 2) * 64;    // wave N origin in block tile
    const int bM   = blockIdx.x * BM;
    const int bN   = blockIdx.y * BN;

    const int kh  = lane >> 4;           // K-half select (ISA 16-bit layouts)
    const int r16 = lane & 15;           // row/col within 16

    v8f acc[2][4];
#pragma unroll
    for (int i = 0; i < 2; ++i)
#pragma unroll
        for (int j = 0; j < 4; ++j) acc[i][j] = (v8f)0.0f;

    // A staging: 128 rows x 8 K-quads of float4; 2 threads/row, 4 quads each
    const int sm  = tid >> 1;                // 0..127 (row)
    const int sq  = (tid & 1) * 4;           // first quad: 0 or 4
    const int gmc = min(bM + sm, M - 1);     // clamp: loads always valid
    const float scA = rdeg[gmc];             // hoisted mean scale (phase 0)

    // B staging: 512 x 16B chunks, exactly two per thread (tid, tid+256)
    const int bn0 = tid >> 2, bc0 = (tid & 3) * 8;             // chunk 0
    const int bn1 = (tid + 256) >> 2, bc1 = bc0;               // chunk 1

#pragma unroll 1
    for (int phase = 0; phase < 2; ++phase) {
        const float* Asrc = phase ? root : agg;
        const float sc    = phase ? 1.0f : scA;
#pragma unroll 1
        for (int kk = 0; kk < DFEAT; kk += BK) {
            const int k0 = phase * DFEAT + kk;   // global K for B

            // ---- B tile: async DMA to LDS (ASYNCcnt) ----
#if HAVE_ASYNC_LDS
            __builtin_amdgcn_global_load_async_to_lds_b128(
                (vi4*)(Wcat + (size_t)(bN + bn0) * KCAT + k0 + bc0),
                (vi4*)&Bs[bn0][bc0], 0, 0);
            __builtin_amdgcn_global_load_async_to_lds_b128(
                (vi4*)(Wcat + (size_t)(bN + bn1) * KCAT + k0 + bc1),
                (vi4*)&Bs[bn1][bc1], 0, 0);
#else
#pragma unroll
            for (int t = tid; t < BN * (BK / 4); t += 256) {
                int n  = t >> 3;
                int kq = t & 7;
                *(h4*)&Bs[n][kq * 4] =
                    *(const h4*)(Wcat + (size_t)(bN + n) * KCAT + k0 + kq * 4);
            }
#endif
            // ---- A tile: convert f32->f16, fuse mean (rdeg) scaling ----
#pragma unroll
            for (int u = 0; u < 4; ++u) {
                int q = sq + u;               // quad index 0..7
                float4 v = *(const float4*)(Asrc + (size_t)gmc * DFEAT +
                                            kk + q * 4);
                h4 hv = { (_Float16)(v.x * sc), (_Float16)(v.y * sc),
                          (_Float16)(v.z * sc), (_Float16)(v.w * sc) };
                *(h4*)&As[sm][q * 4] = hv;
            }
            if (k0 + BK < KCAT)                   // -> global_prefetch_b8
                __builtin_prefetch(
                    Wcat + (size_t)(bN + (tid >> 1)) * KCAT + k0 + BK, 0, 1);
#if HAVE_ASYNC_LDS
            wait_async_zero();                    // drain this wave's DMA
#endif
            __syncthreads();

            // ---- fragments per ISA 7.12.2 16-bit layouts, 1 WMMA k-step ----
            v16h af[2];
#pragma unroll
            for (int mi = 0; mi < 2; ++mi) {
                int m = wm + mi * 16 + r16;
#pragma unroll
                for (int v = 0; v < 8; ++v) {
                    int kb = ((v < 4) ? 2 * v : 16 + 2 * (v - 4)) + kh * 8;
                    h2 p = *(const h2*)&As[m][kb];
                    af[mi][2 * v]     = p[0];
                    af[mi][2 * v + 1] = p[1];
                }
            }
#pragma unroll
            for (int ni = 0; ni < 4; ++ni) {
                v16h bf;
                int n = wn + ni * 16 + r16;
#pragma unroll
                for (int v = 0; v < 8; ++v) {
                    int kb = ((v < 4) ? 2 * v : 16 + 2 * (v - 4)) + kh * 8;
                    h2 p = *(const h2*)&Bs[n][kb];
                    bf[2 * v]     = p[0];
                    bf[2 * v + 1] = p[1];
                }
                acc[0][ni] = __builtin_amdgcn_wmma_f32_16x16x32_f16(
                    false, af[0], false, bf, (short)0, acc[0][ni], false, false);
                acc[1][ni] = __builtin_amdgcn_wmma_f32_16x16x32_f16(
                    false, af[1], false, bf, (short)0, acc[1][ni], false, false);
            }
            __syncthreads();
        }
    }

    // ---- epilogue: bias (+ReLU), f32 store.  C layout: M = r + 8*kh ----
#pragma unroll
    for (int ni = 0; ni < 4; ++ni) {
        int gn = bN + wn + ni * 16 + r16;
        float bv = bias[gn];
#pragma unroll
        for (int mi = 0; mi < 2; ++mi) {
#pragma unroll
            for (int r = 0; r < 8; ++r) {
                int gm = bM + wm + mi * 16 + r + kh * 8;
                if (gm < M) {
                    float val = acc[mi][ni][r] + bv;
                    if (do_relu) val = fmaxf(val, 0.0f);
                    out[(size_t)gm * Nout + gn] = val;
                }
            }
        }
    }
}

// ---------------------------------------------------------------------------
// In-place log_softmax over 256 columns; one wave32 per row.
// ---------------------------------------------------------------------------
__global__ void sage_log_softmax(float* __restrict__ out, int n, int cols) {
    int wave = (blockIdx.x * blockDim.x + threadIdx.x) >> 5;
    int lane = threadIdx.x & 31;
    if (wave >= n) return;
    float* p = out + (size_t)wave * cols;
    float v[8];
    float mx = -3.402823466e+38f;
#pragma unroll
    for (int j = 0; j < 8; ++j) {
        v[j] = p[lane + j * 32];
        mx = fmaxf(mx, v[j]);
    }
#pragma unroll
    for (int off = 16; off > 0; off >>= 1)
        mx = fmaxf(mx, __shfl_xor(mx, off, 32));
    float s = 0.0f;
#pragma unroll
    for (int j = 0; j < 8; ++j) s += __expf(v[j] - mx);
#pragma unroll
    for (int off = 16; off > 0; off >>= 1)
        s += __shfl_xor(s, off, 32);
    float l = __logf(s);
#pragma unroll
    for (int j = 0; j < 8; ++j) p[lane + j * 32] = v[j] - mx - l;
}

// ---------------------------------------------------------------------------
static inline size_t align256(size_t x) { return (x + 255) & ~(size_t)255; }

extern "C" void kernel_launch(void* const* d_in, const int* in_sizes, int n_in,
                              void* d_out, int out_size, void* d_ws, size_t ws_size,
                              hipStream_t stream) {
    const float*     x   = (const float*)d_in[0];
    const long long* e1  = (const long long*)d_in[1];
    const long long* e2  = (const long long*)d_in[2];
    const float*     W1l = (const float*)d_in[3];
    const float*     b1  = (const float*)d_in[4];
    const float*     W1r = (const float*)d_in[5];
    const float*     W2l = (const float*)d_in[6];
    const float*     b2  = (const float*)d_in[7];
    const float*     W2r = (const float*)d_in[8];
    float* out = (float*)d_out;

    const int N = NNODES, E = NEDGES;
    const size_t featB = (size_t)N * DFEAT * sizeof(float);

    char* ws = (char*)d_ws;
    float*     agg  = (float*)ws;               ws += align256(featB);
    float*     h    = (float*)ws;               ws += align256(featB);
    float*     deg  = (float*)ws;               ws += align256((size_t)N * 4);
    float*     rdeg = (float*)ws;               ws += align256((size_t)N * 4);
    _Float16*  W1c  = (_Float16*)ws;            ws += align256((size_t)512 * KCAT * 2);
    _Float16*  W2c  = (_Float16*)ws;            ws += align256((size_t)256 * KCAT * 2);

    // pack weights to f16 [Nout x 1024]
    {
        int t1 = 512 * KCAT, t2 = 256 * KCAT;
        sage_pack_w<<<(t1 + 255) / 256, 256, 0, stream>>>(W1l, W1r, W1c, 512);
        sage_pack_w<<<(t2 + 255) / 256, 256, 0, stream>>>(W2l, W2r, W2c, 256);
    }

    dim3 blk(256);
    const int scatterBlocks = (E * 32 + 255) / 256;      // one wave per edge
    const int gemmMBlocks   = (N + BM - 1) / BM;         // 391

    // ---------------- layer 1 ----------------
    (void)hipMemsetAsync(agg, 0, featB, stream);
    (void)hipMemsetAsync(deg, 0, (size_t)N * 4, stream);
    sage_scatter<<<scatterBlocks, blk, 0, stream>>>(x, e1, agg, deg, E);
    sage_rdeg<<<(N + 255) / 256, blk, 0, stream>>>(deg, rdeg, N);
    {
        dim3 grid(gemmMBlocks, 512 / BN);                // (391, 4)
        sage_gemm<<<grid, blk, 0, stream>>>(agg, rdeg, x, W1c, b1, h,
                                            N, 512, /*relu=*/1);
    }

    // ---------------- layer 2 ----------------
    (void)hipMemsetAsync(agg, 0, featB, stream);
    (void)hipMemsetAsync(deg, 0, (size_t)N * 4, stream);
    sage_scatter<<<scatterBlocks, blk, 0, stream>>>(h, e2, agg, deg, E);
    sage_rdeg<<<(N + 255) / 256, blk, 0, stream>>>(deg, rdeg, N);
    {
        dim3 grid(gemmMBlocks, 256 / BN);                // (391, 2)
        sage_gemm<<<grid, blk, 0, stream>>>(agg, rdeg, h, W2c, b2, out,
                                            N, 256, /*relu=*/0);
    }

    // ---------------- log_softmax (in place) ----------------
    sage_log_softmax<<<(N * 32 + 255) / 256, blk, 0, stream>>>(out, N, 256);
}
